// GlobalModel_55800215110033
// MI455X (gfx1250) — matrix operand order repaired and checked by reading, not compiled, not run on hardware
//
#include <hip/hip_runtime.h>
#include <initializer_list>

// ---------------------------------------------------------------------------
// Problem dimensions (fixed by the reference)
// ---------------------------------------------------------------------------
#define BB   64    // batch
#define TT   256   // mel timesteps
#define UU   256   // text length
#define NMEL 80
#define PRE  128
#define EMB  15
#define NF   128
#define ENC  128
#define DEC  512
#define WMX  10    // attention mixture components

typedef __attribute__((ext_vector_type(16))) __bf16 v16bf;
typedef __attribute__((ext_vector_type(8)))  __bf16 v8bf;
typedef __attribute__((ext_vector_type(8)))  float  v8f;

// ---------------------------------------------------------------------------
// Generic bf16-WMMA GEMM:  C[M,N] = act( concat-A[M,K] @ W[K,N] + bias (+resid) )
//   * A f32, assembled from up to 4 segments (ptr,row-stride,width) or im2col
//     conv addressing.  REQUIREMENT: every segment prefix-sum is a multiple of
//     32 (or there is a single segment), so a 32-wide K-chunk never straddles
//     a boundary -> segment choice is uniform per chunk (scalar branches only).
//   * W bf16 row-major with leading dim w_ld (>=N, multiple of 4).
//   * 256 threads = 8 wave32; block tile 64x64; wave = one 16x16 m-tile x two
//     n-tiles; LDS-staged K-chunks of 32; B staged TRANSPOSED so both A and B
//     fragments are contiguous 16B runs -> ds_load_b128.
// ---------------------------------------------------------------------------
struct Seg { const float* p; long rs; int w; };
struct GemmArgs {
  Seg seg[4];
  int nseg;
  const __bf16* W;        // K x w_ld, bf16
  int w_ld;
  const float* bias;      // N (or null)
  const float* resid; long resid_rs;   // residual add AFTER relu (conv stacks)
  float* C;  long c_rs;
  float* C2; long c2_rs;  // optional second destination (ctx write)
  int M, N, K;
  int relu;
  int conv_cin, conv_U;   // conv_cin>0 selects im2col A addressing
  const float* conv_x;    // (B,U,cin) input for conv mode
};

#define GBK 32
#define LDP 40   // LDS row stride in bf16 (80B = 5x16B, keeps 16B alignment)

__global__ __launch_bounds__(256) void gemm_wmma_kernel(GemmArgs g) {
  __shared__ alignas(16) __bf16 As[64][LDP];   // (m, k)
  __shared__ alignas(16) __bf16 Bt[64][LDP];   // (n, k)  -- transposed weights

  const int m0 = blockIdx.x * 64, n0 = blockIdx.y * 64;
  const int tid  = threadIdx.x;
  const int lane = tid & 31;
  const int wave = tid >> 5;
  const int half = lane >> 4;     // 0: lanes 0-15, 1: lanes 16-31
  const int lm   = lane & 15;
  const int mt   = wave & 3;      // m-tile 0..3
  const int np   = wave >> 2;     // n-pair 0..1 (n-tiles 2np, 2np+1)

  v8f acc0 = {}; v8f acc1 = {};

  for (int kc = 0; kc < g.K; kc += GBK) {
    // ---- uniform (per-chunk) segment selection: scalar control flow ----
    const float* abase = nullptr; long ars = 0; int aw = 0;
    if (g.conv_cin == 0) {
      int rem = kc;
      for (int s = 0; s < g.nseg; ++s) {
        if (rem < g.seg[s].w) { abase = g.seg[s].p + rem; ars = g.seg[s].rs;
                                aw = g.seg[s].w - rem; break; }
        rem -= g.seg[s].w;
      }
    }
    __syncthreads();
    // ---- stage A chunk (64 x 32), f32 -> bf16, coalesced along k ----
#pragma unroll
    for (int i = 0; i < 8; ++i) {
      int flat = tid + i * 256;
      int m = flat >> 5, k = flat & 31;
      int gm = m0 + m;
      float v = 0.f;
      if (g.conv_cin > 0) {
        int gk = kc + k;
        if (gm < g.M && gk < g.K) {
          int cin = g.conv_cin, Uc = g.conv_U;
          int b = gm / Uc, u = gm % Uc;
          int tap = gk / cin, ci = gk % cin;
          int uu = u + tap - 2;                      // SAME padding, width 5
          if (uu >= 0 && uu < Uc)
            v = g.conv_x[((long)b * Uc + uu) * cin + ci];
        }
      } else if (gm < g.M && k < aw) {
        v = abase[(long)gm * ars + k];
      }
      As[m][k] = (__bf16)v;
    }
    // ---- stage B chunk (32 x 64) TRANSPOSED into Bt[n][k] ----
#pragma unroll
    for (int i = 0; i < 8; ++i) {
      int flat = tid + i * 256;
      int k = flat >> 6, n = flat & 63;
      int gk = kc + k, gn = n0 + n;
      __bf16 v = (__bf16)0.f;
      if (gk < g.K && gn < g.N) v = g.W[(long)gk * g.w_ld + gn];
      Bt[n][k] = v;
    }
    // prefetch next weight chunk (emits global_prefetch_b8)
    if (tid == 0 && kc + GBK < g.K)
      __builtin_prefetch(g.W + (long)(kc + GBK) * g.w_ld + n0, 0, 1);
    __syncthreads();

    // ---- fragments (ISA 7.12.2): contiguous 16B runs -> ds_load_b128 ----
    // A: lane holds row m=lm, K = 16*half + e
    const v8bf* arow = (const v8bf*)&As[mt * 16 + lm][0];
    v8bf a_lo = arow[half * 2];      // K = 16*half + 0..7
    v8bf a_hi = arow[half * 2 + 1];  // K = 16*half + 8..15
    v16bf af = __builtin_shufflevector(a_lo, a_hi,
        0,1,2,3,4,5,6,7,8,9,10,11,12,13,14,15);
    // B: lane holds col n, K = 16*half + e (contiguous 32B run)
    const v8bf* br0 = (const v8bf*)&Bt[np * 32 + lm][0];
    v16bf bf0 = __builtin_shufflevector(br0[half * 2], br0[half * 2 + 1],
        0,1,2,3,4,5,6,7,8,9,10,11,12,13,14,15);
    const v8bf* br1 = (const v8bf*)&Bt[np * 32 + 16 + lm][0];
    v16bf bf1 = __builtin_shufflevector(br1[half * 2], br1[half * 2 + 1],
        0,1,2,3,4,5,6,7,8,9,10,11,12,13,14,15);

    acc0 = __builtin_amdgcn_wmma_f32_16x16x32_bf16(false, af, false, bf0,
                                                   (short)0, acc0, false, false);
    acc1 = __builtin_amdgcn_wmma_f32_16x16x32_bf16(false, af, false, bf1,
                                                   (short)0, acc1, false, false);
  }

  // ---- epilogue: C/D layout M = 8*half + r, N = lane%16 ----
#pragma unroll
  for (int r = 0; r < 8; ++r) {
    int gm = m0 + mt * 16 + half * 8 + r;
    if (gm >= g.M) continue;
#pragma unroll
    for (int t = 0; t < 2; ++t) {
      int gn = n0 + np * 32 + t * 16 + lm;
      if (gn >= g.N) continue;
      float v = t ? acc1[r] : acc0[r];
      if (g.bias)  v += g.bias[gn];
      if (g.relu)  v = fmaxf(v, 0.f);                 // relu BEFORE residual (ref)
      if (g.resid) v += g.resid[(long)gm * g.resid_rs + gn];
      g.C[(long)gm * g.c_rs + gn] = v;
      if (g.C2) g.C2[(long)gm * g.c2_rs + gn] = v;
    }
  }
}

// ---------------------------------------------------------------------------
// Elementwise helpers
// ---------------------------------------------------------------------------
__device__ __forceinline__ float sigmoidf_(float x) { return 1.f / (1.f + __expf(-x)); }
__device__ __forceinline__ float softplusf_(float x) {
  return (x > 20.f) ? x : log1pf(__expf(x));
}

// LSTM cell with mask blending: h,c updated in place; optional copy of new h.
__global__ void lstm_cell_kernel(const float* z, float* h, float* c,
                                 const float* mask,   // per-batch-row, or null
                                 float* hcopy, long hcopy_rs,
                                 int M, int H) {
  long idx = (long)blockIdx.x * blockDim.x + threadIdx.x;
  if (idx >= (long)M * H) return;
  int b = (int)(idx / H), j = (int)(idx % H);
  const float* zr = z + (long)b * 4 * H;
  float zi = zr[j], zf = zr[H + j], zg = zr[2 * H + j], zo = zr[3 * H + j];
  float co = c[idx];
  float cn = sigmoidf_(zf) * co + sigmoidf_(zi) * tanhf(zg);
  float hn = sigmoidf_(zo) * tanhf(cn);
  float m = mask ? mask[b] : 1.f;
  float ho = h[idx];
  hn = m * hn + (1.f - m) * ho;
  cn = m * cn + (1.f - m) * co;
  h[idx] = hn; c[idx] = cn;
  if (hcopy) hcopy[(long)b * hcopy_rs + j] = hn;
}

// GMM attention step: one block per batch row (256 threads, U==256, 2*ENC==256)
__global__ void attention_kernel(const float* pr,        // B x 30 (a|b|k)
                                 float* att_w, float* att_k,
                                 const float* ctx,       // (U,B,2*ENC)
                                 const float* text_mask, // (U,B)
                                 const float* mel_mask_t // B (row t)
                                 ) {
  int b = blockIdx.x, tid = threadIdx.x;
  __shared__ float a_s[WMX], b_s[WMX], k_s[WMX];
  __shared__ float phi[UU];
  if (tid < WMX) {
    float ah = pr[b * 3 * WMX + tid];
    float bh = pr[b * 3 * WMX + WMX + tid];
    float kh = pr[b * 3 * WMX + 2 * WMX + tid];
    a_s[tid] = softplusf_(ah);
    b_s[tid] = softplusf_(bh);
    k_s[tid] = att_k[b * WMX + tid] + softplusf_(kh);
  }
  __syncthreads();
  {
    float u = (float)tid, acc = 0.f;
#pragma unroll
    for (int w = 0; w < WMX; ++w) {
      float d = k_s[w] - u;
      acc += a_s[w] * __expf(-b_s[w] * d * d);
    }
    phi[tid] = acc * text_mask[(long)tid * BB + b];   // phi_mask = text_mask.T
  }
  __syncthreads();
  float wt = 0.f;
  for (int u = 0; u < UU; ++u)
    wt += phi[u] * ctx[((long)u * BB + b) * (2 * ENC) + tid];
  float m = mel_mask_t[b];
  long iw = (long)b * (2 * ENC) + tid;
  att_w[iw] = m * wt + (1.f - m) * att_w[iw];
  if (tid < WMX) {
    long ik = (long)b * WMX + tid;
    att_k[ik] = m * k_s[tid] + (1.f - m) * att_k[ik];
  }
}

__global__ void embed_kernel(const int* text, const float* emb, float* x) {
  long idx = (long)blockIdx.x * blockDim.x + threadIdx.x;
  long n = (long)BB * UU * EMB;
  if (idx >= n) return;
  int e = (int)(idx % EMB);
  long r = idx / EMB;
  int u = (int)(r % UU);
  int b = (int)(r / UU);
  x[idx] = emb[(long)text[(long)u * BB + b] * EMB + e];
}

__global__ void cvt_bf16_kernel(const float* src, __bf16* dst, long n) {
  long i = (long)blockIdx.x * blockDim.x + threadIdx.x;
  if (i < n) dst[i] = (__bf16)src[i];
}

// f32 (K x N) -> bf16 (K x ldN) with zero column padding.
__global__ void cvt_bf16_pad_kernel(const float* src, __bf16* dst,
                                    int K, int N, int ldN) {
  long i = (long)blockIdx.x * blockDim.x + threadIdx.x;
  long n = (long)K * ldN;
  if (i >= n) return;
  int col = (int)(i % ldN);
  long row = i / ldN;
  dst[i] = (__bf16)((col < N) ? src[row * N + col] : 0.f);
}

// Collapse the 3 masked kernels (widths 1/3/5) of one conv stack into a single
// 5-tap kernel (bf16) + combined bias (f32).  w: (3,5,cin,128), b: (3,128).
__global__ void conv_combine_kernel(const float* w, const float* b,
                                    __bf16* wc, float* bc, int cin) {
  long n = 5L * cin * NF;
  long idx = (long)blockIdx.x * blockDim.x + threadIdx.x;
  if (idx < n) {
    int co = (int)(idx % NF);
    long r = idx / NF;
    int ci = (int)(r % cin);
    int tap = (int)(r / cin);
    float v = 0.f;
    if (tap == 2)              v += w[((0L * 5 + tap) * cin + ci) * NF + co]; // k=1
    if (tap >= 1 && tap <= 3)  v += w[((1L * 5 + tap) * cin + ci) * NF + co]; // k=3
    v += w[((2L * 5 + tap) * cin + ci) * NF + co];                            // k=5
    wc[idx] = (__bf16)v;
  }
  if (idx < NF) bc[idx] = b[idx] + b[NF + idx] + b[2 * NF + idx];
}

__global__ void zero_kernel(float* p, long n) {
  long i = (long)blockIdx.x * blockDim.x + threadIdx.x;
  if (i < n) p[i] = 0.f;
}

// ---------------------------------------------------------------------------
// Host orchestration
// ---------------------------------------------------------------------------
extern "C" void kernel_launch(void* const* d_in, const int* in_sizes, int n_in,
                              void* d_out, int out_size, void* d_ws, size_t ws_size,
                              hipStream_t stream) {
  const float* in_mels   = (const float*)d_in[0];   // (T,B,NMEL)
  const float* mel_mask  = (const float*)d_in[1];   // (T,B)
  const int*   text      = (const int*)  d_in[2];   // (U,B)
  const float* text_mask = (const float*)d_in[3];   // (U,B)
  const float* w_pre1 = (const float*)d_in[4];  const float* b_pre1 = (const float*)d_in[5];
  const float* w_pre2 = (const float*)d_in[6];  const float* b_pre2 = (const float*)d_in[7];
  const float* emb    = (const float*)d_in[8];
  const float* cw0 = (const float*)d_in[9];   const float* cb0 = (const float*)d_in[10];
  const float* cw1 = (const float*)d_in[11];  const float* cb1 = (const float*)d_in[12];
  const float* cw2 = (const float*)d_in[13];  const float* cb2 = (const float*)d_in[14];
  const float* w_bif = (const float*)d_in[15]; const float* b_bif = (const float*)d_in[16];
  const float* w_bib = (const float*)d_in[17]; const float* b_bib = (const float*)d_in[18];
  const float* w_att = (const float*)d_in[19]; const float* b_att = (const float*)d_in[20];
  const float* w_ap  = (const float*)d_in[21]; const float* b_ap  = (const float*)d_in[22];
  const float* w_r1  = (const float*)d_in[23]; const float* b_r1  = (const float*)d_in[24];
  const float* w_r2  = (const float*)d_in[25]; const float* b_r2  = (const float*)d_in[26];
  const float* w_out = (const float*)d_in[27]; const float* b_out = (const float*)d_in[28];
  float* out = (float*)d_out;                        // (T,B,NMEL)

  // ---- bump allocator over d_ws (total ~67 MB) ----
  char* ws = (char*)d_ws;
  size_t off = 0;
  auto alloc_f = [&](size_t n) -> float*  { off = (off + 255) & ~(size_t)255; float*  p = (float*) (ws + off); off += n * 4; return p; };
  auto alloc_h = [&](size_t n) -> __bf16* { off = (off + 255) & ~(size_t)255; __bf16* p = (__bf16*)(ws + off); off += n * 2; return p; };

  // bf16 weights (~16 MB)
  __bf16* hpre1 = alloc_h((size_t)NMEL * PRE);
  __bf16* hpre2 = alloc_h((size_t)PRE * PRE);
  __bf16* hc0   = alloc_h(5L * EMB * NF);
  __bf16* hc1   = alloc_h(5L * NF * NF);
  __bf16* hc2   = alloc_h(5L * NF * NF);
  __bf16* hbif  = alloc_h((size_t)(NF + ENC) * 4 * ENC);
  __bf16* hbib  = alloc_h((size_t)(NF + ENC) * 4 * ENC);
  __bf16* hatt  = alloc_h((size_t)(PRE + 2 * ENC + DEC) * 4 * DEC);
  __bf16* hap   = alloc_h((size_t)DEC * 32);           // 30 cols padded to 32
  __bf16* hr1   = alloc_h((size_t)(PRE + 2 * ENC + 2 * DEC) * 4 * DEC);
  __bf16* hr2   = alloc_h((size_t)(PRE + 2 * ENC + 2 * DEC) * 4 * DEC);
  __bf16* hout  = alloc_h((size_t)DEC * NMEL);

  // f32 buffers (~50 MB)
  float* cbias0 = alloc_f(NF); float* cbias1 = alloc_f(NF); float* cbias2 = alloc_f(NF);
  float* tmp    = alloc_f((size_t)TT * BB * PRE);      // prenet intermediate
  float* corr   = alloc_f((size_t)TT * BB * PRE);      // (T,B,PRE)
  float* xemb   = alloc_f((size_t)BB * UU * EMB);      // (B,U,EMB)
  float* cbuf0  = alloc_f((size_t)BB * UU * NF);       // conv ping
  float* cbuf1  = alloc_f((size_t)BB * UU * NF);       // conv pong
  float* ctx    = alloc_f((size_t)UU * BB * 2 * ENC);  // (U,B,2*ENC)
  float* zf     = alloc_f((size_t)BB * 4 * ENC);
  float* zb     = alloc_f((size_t)BB * 4 * ENC);
  float* zbuf   = alloc_f((size_t)BB * 4 * DEC);
  float* prbuf  = alloc_f((size_t)BB * 3 * WMX);
  // contiguous state block (zeroed every call -> deterministic)
  size_t nstate = (size_t)BB * (ENC * 4 + 2 * ENC + WMX + DEC * 6);
  float* st = alloc_f(nstate);
  float* h_f = st;             float* c_f = h_f + BB * ENC;
  float* h_b = c_f + BB * ENC; float* c_b = h_b + BB * ENC;
  float* att_w = c_b + BB * ENC;         // B x 2*ENC
  float* att_k = att_w + BB * 2 * ENC;   // B x WM
  float* att_h = att_k + BB * WMX;  float* att_c = att_h + BB * DEC;
  float* h1 = att_c + BB * DEC;     float* c1 = h1 + BB * DEC;
  float* h2 = c1 + BB * DEC;        float* c2 = h2 + BB * DEC;
  (void)ws_size; (void)in_sizes; (void)n_in; (void)out_size;

  auto cvt = [&](const float* s, __bf16* d, long n) {
    cvt_bf16_kernel<<<dim3((unsigned)((n + 255) / 256)), dim3(256), 0, stream>>>(s, d, n);
  };
  auto gemm = [&](std::initializer_list<Seg> segs, const __bf16* W, const float* bias,
                  float* C, long c_rs, int M, int N, int K, int w_ld,
                  float* C2 = nullptr, long c2_rs = 0) {
    GemmArgs a{}; int i = 0;
    for (auto& s : segs) a.seg[i++] = s;
    a.nseg = i; a.W = W; a.w_ld = w_ld; a.bias = bias;
    a.resid = nullptr; a.resid_rs = 0;
    a.C = C; a.c_rs = c_rs; a.C2 = C2; a.c2_rs = c2_rs;
    a.M = M; a.N = N; a.K = K; a.relu = 0;
    a.conv_cin = 0; a.conv_U = 0; a.conv_x = nullptr;
    dim3 g((M + 63) / 64, (N + 63) / 64);
    gemm_wmma_kernel<<<g, dim3(256), 0, stream>>>(a);
  };
  auto conv = [&](const float* x, int cin, const __bf16* W, const float* bias,
                  float* C, const float* resid) {
    GemmArgs a{}; a.nseg = 0; a.W = W; a.w_ld = NF; a.bias = bias;
    a.resid = resid; a.resid_rs = NF;
    a.C = C; a.c_rs = NF; a.C2 = nullptr; a.c2_rs = 0;
    a.M = BB * UU; a.N = NF; a.K = 5 * cin; a.relu = 1;
    a.conv_cin = cin; a.conv_U = UU; a.conv_x = x;
    dim3 g((a.M + 63) / 64, (a.N + 63) / 64);
    gemm_wmma_kernel<<<g, dim3(256), 0, stream>>>(a);
  };
  auto cell = [&](const float* z, float* h, float* c, const float* mask,
                  float* hcopy, long hcopy_rs, int H) {
    long n = (long)BB * H;
    lstm_cell_kernel<<<dim3((unsigned)((n + 255) / 256)), dim3(256), 0, stream>>>(
        z, h, c, mask, hcopy, hcopy_rs, BB, H);
  };

  // ---- weight conversion / conv-kernel collapse ----
  cvt(w_pre1, hpre1, (long)NMEL * PRE);
  cvt(w_pre2, hpre2, (long)PRE * PRE);
  cvt(w_bif, hbif, (long)(NF + ENC) * 4 * ENC);
  cvt(w_bib, hbib, (long)(NF + ENC) * 4 * ENC);
  cvt(w_att, hatt, (long)(PRE + 2 * ENC + DEC) * 4 * DEC);
  cvt(w_r1,  hr1,  (long)(PRE + 2 * ENC + 2 * DEC) * 4 * DEC);
  cvt(w_r2,  hr2,  (long)(PRE + 2 * ENC + 2 * DEC) * 4 * DEC);
  cvt(w_out, hout, (long)DEC * NMEL);
  cvt_bf16_pad_kernel<<<dim3((DEC * 32 + 255) / 256), dim3(256), 0, stream>>>(
      w_ap, hap, DEC, 3 * WMX, 32);
  conv_combine_kernel<<<dim3((5 * EMB * NF + 255) / 256), dim3(256), 0, stream>>>(cw0, cb0, hc0, cbias0, EMB);
  conv_combine_kernel<<<dim3((5 * NF * NF + 255) / 256), dim3(256), 0, stream>>>(cw1, cb1, hc1, cbias1, NF);
  conv_combine_kernel<<<dim3((5 * NF * NF + 255) / 256), dim3(256), 0, stream>>>(cw2, cb2, hc2, cbias2, NF);
  zero_kernel<<<dim3((unsigned)((nstate + 255) / 256)), dim3(256), 0, stream>>>(st, (long)nstate);

  // ---- prenet: corr = (in_mels @ W1 + b1) @ W2 + b2 ----
  gemm({Seg{in_mels, NMEL, NMEL}}, hpre1, b_pre1, tmp, PRE, TT * BB, PRE, NMEL, PRE);
  gemm({Seg{tmp, PRE, PRE}},       hpre2, b_pre2, corr, PRE, TT * BB, PRE, PRE, PRE);

  // ---- text embedding + conv stacks ----
  {
    long n = (long)BB * UU * EMB;
    embed_kernel<<<dim3((unsigned)((n + 255) / 256)), dim3(256), 0, stream>>>(text, emb, xemb);
  }
  conv(xemb,  EMB, hc0, cbias0, cbuf0, nullptr);  // stack 0, no residual
  conv(cbuf0, NF,  hc1, cbias1, cbuf1, cbuf0);    // stack 1, +x
  conv(cbuf1, NF,  hc2, cbias2, cbuf0, cbuf1);    // stack 2, +x  -> conv_text
  float* ct = cbuf0;                               // (B,U,NF)

  // ---- bidirectional LSTM encoder; h written straight into ctx halves ----
  for (int u = 0; u < UU; ++u) {
    gemm({Seg{ct + (long)u * NF, (long)UU * NF, NF}, Seg{h_f, ENC, ENC}},
         hbif, b_bif, zf, 4 * ENC, BB, 4 * ENC, NF + ENC, 4 * ENC);
    cell(zf, h_f, c_f, text_mask + (long)u * BB,
         ctx + (long)u * BB * 2 * ENC, 2 * ENC, ENC);
    int ub = UU - 1 - u;
    gemm({Seg{ct + (long)ub * NF, (long)UU * NF, NF}, Seg{h_b, ENC, ENC}},
         hbib, b_bib, zb, 4 * ENC, BB, 4 * ENC, NF + ENC, 4 * ENC);
    cell(zb, h_b, c_b, text_mask + (long)ub * BB,
         ctx + (long)ub * BB * 2 * ENC + ENC, 2 * ENC, ENC);
  }

  // ---- decoder scan (256 sequential steps) ----
  for (int t = 0; t < TT; ++t) {
    const float* corr_t = corr + (long)t * BB * PRE;       // rows stride PRE
    const float* m_t = mel_mask + (long)t * BB;
    // attention LSTM: z = [corr_t | att_w | att_h] @ W_att + b
    gemm({Seg{corr_t, PRE, PRE}, Seg{att_w, 2 * ENC, 2 * ENC}, Seg{att_h, DEC, DEC}},
         hatt, b_att, zbuf, 4 * DEC, BB, 4 * DEC, PRE + 2 * ENC + DEC, 4 * DEC);
    cell(zbuf, att_h, att_c, m_t, nullptr, 0, DEC);        // mask==1 => same as ref
    // attention projection + GMM attention
    gemm({Seg{att_h, DEC, DEC}}, hap, b_ap, prbuf, 3 * WMX, BB, 3 * WMX, DEC, 32);
    attention_kernel<<<dim3(BB), dim3(UU), 0, stream>>>(prbuf, att_w, att_k, ctx,
                                                        text_mask, m_t);
    // rnn1: z = [corr_t | w_t | ah | h1] @ W1 + b1
    gemm({Seg{corr_t, PRE, PRE}, Seg{att_w, 2 * ENC, 2 * ENC},
          Seg{att_h, DEC, DEC}, Seg{h1, DEC, DEC}},
         hr1, b_r1, zbuf, 4 * DEC, BB, 4 * DEC, PRE + 2 * ENC + 2 * DEC, 4 * DEC);
    cell(zbuf, h1, c1, m_t, nullptr, 0, DEC);
    // rnn2: z = [corr_t | w_t | h1 | h2] @ W2 + b2
    gemm({Seg{corr_t, PRE, PRE}, Seg{att_w, 2 * ENC, 2 * ENC},
          Seg{h1, DEC, DEC}, Seg{h2, DEC, DEC}},
         hr2, b_r2, zbuf, 4 * DEC, BB, 4 * DEC, PRE + 2 * ENC + 2 * DEC, 4 * DEC);
    cell(zbuf, h2, c2, m_t, nullptr, 0, DEC);
    // fused output projection: out[t] = h2 @ W_out + b_out  (avoids 32MB buffer)
    gemm({Seg{h2, DEC, DEC}}, hout, b_out, out + (long)t * BB * NMEL, NMEL,
         BB, NMEL, DEC, NMEL);
  }
}
